// GNN_Backbone_60026462929460
// MI455X (gfx1250) — matrix operand order repaired and checked by reading, not compiled, not run on hardware
//
#include <hip/hip_runtime.h>

// ---------------- problem constants ----------------
#define NN   100000
#define EE   600000
#define HID  128
#define HID2 256
#define NLAY 2
#define BN_EPS 1e-5f

typedef __attribute__((ext_vector_type(16))) _Float16 v16h;
typedef __attribute__((ext_vector_type(8)))  _Float16 v8h;
typedef __attribute__((ext_vector_type(8)))  float    v8f;
typedef __attribute__((ext_vector_type(4)))  unsigned int u32x4;
typedef __attribute__((ext_vector_type(4)))  int          i32x4;
typedef __attribute__((ext_vector_type(8)))  int          i32x8;

// ---------------- Tensor Data Mover: stage a contiguous 2D f16 matrix into LDS ----
// Loads `rows` rows of `row_qw` 8-byte quads each (contiguous, stride == row_qw)
// from global memory into LDS at byte offset lds_off.  D# layout per CDNA5 ISA §8.
__device__ __forceinline__ void tdm_load_2d(unsigned int lds_off, const void* gptr,
                                            unsigned int row_qw, unsigned int rows)
{
    unsigned long long ga = (unsigned long long)(size_t)gptr;
    u32x4 g0;
    g0[0] = 1u;                                              // count=1 (valid), user mode
    g0[1] = lds_off;                                         // lds_addr (bytes)
    g0[2] = (unsigned int)(ga & 0xffffffffu);                // global_addr[31:0]
    g0[3] = (unsigned int)((ga >> 32) & 0x01ffffffu)         // global_addr[56:32]
          | (2u << 30);                                      // type = 2 ("image")
    i32x8 g1;
    g1[0] = (int)(3u << 16);                                 // data_size = 8 bytes
    g1[1] = (int)((row_qw & 0xffffu) << 16);                 // tensor_dim0[15:0]
    g1[2] = (int)(((row_qw >> 16) & 0xffffu)                 // tensor_dim0[31:16]
          | ((rows & 0xffffu) << 16));                       // tensor_dim1[15:0]
    g1[3] = (int)(((rows >> 16) & 0xffffu)                   // tensor_dim1[31:16]
          | ((row_qw & 0xffffu) << 16));                     // tile_dim0 = row_qw
    g1[4] = (int)(rows & 0xffffu);                           // tile_dim1 = rows, tile_dim2=0
    g1[5] = (int)row_qw;                                     // tensor_dim0_stride[31:0]
    g1[6] = 0;                                               // stride hi / dim1_stride lo
    g1[7] = 0;
    i32x4 z4 = {0, 0, 0, 0};
#if defined(__clang_major__) && (__clang_major__ >= 23)
    i32x8 z8 = {0, 0, 0, 0, 0, 0, 0, 0};
    __builtin_amdgcn_tensor_load_to_lds(g0, g1, z4, z4, z8, 0);
#else
    __builtin_amdgcn_tensor_load_to_lds(g0, g1, z4, z4, 0);
#endif
}

// ---------------- tiny utility kernels ----------------
__global__ __launch_bounds__(256) void k_zero(float* __restrict__ p, long n) {
    long i = (long)blockIdx.x * blockDim.x + threadIdx.x;
    long stride = (long)gridDim.x * blockDim.x;
    for (; i < n; i += stride) p[i] = 0.0f;
}

__global__ __launch_bounds__(256) void k_cvt_f16(const float* __restrict__ s,
                                                 _Float16* __restrict__ d, long n) {
    long i = (long)blockIdx.x * blockDim.x + threadIdx.x;
    long stride = (long)gridDim.x * blockDim.x;
    for (; i < n; i += stride) d[i] = (_Float16)s[i];
}

// src is [K x Ncol] row-major fp32 -> dst is [Ncol x K] f16 (transposed)
__global__ __launch_bounds__(256) void k_transpose_cvt(const float* __restrict__ s,
                                                       _Float16* __restrict__ d,
                                                       int K, int Ncol) {
    int i = blockIdx.x * blockDim.x + threadIdx.x;
    int total = K * Ncol;
    if (i >= total) return;
    int k = i / Ncol, n = i - k * Ncol;
    d[n * K + k] = (_Float16)s[i];
}

// ---------------- WMMA GEMM: C[M,Ncol] = A[M,K](f16) @ Bt^T + bias ----------------
// Bt is [Ncol x K] f16 (weights transposed), staged into LDS via TDM once per block.
// 8 waves/block, 16 rows per wave.
__global__ __launch_bounds__(256) void k_gemm_f16(
    const _Float16* __restrict__ A, const _Float16* __restrict__ Bt,
    const float* __restrict__ bias, float* __restrict__ C,
    int M, int K, int Ncol, int relu)
{
    __shared__ __attribute__((aligned(16))) _Float16 sB[32768];   // 64 KB max (Ncol*K)

    const int lane = threadIdx.x & 31;
    const int wave = threadIdx.x >> 5;
    const int m0 = (blockIdx.x * 8 + wave) * 16;

    // Stage the whole weight matrix into LDS with the Tensor Data Mover.
    if (wave == 0) {
        tdm_load_2d((unsigned int)(size_t)&sB[0], Bt,
                    (unsigned int)(K / 4), (unsigned int)Ncol);
        __builtin_amdgcn_s_wait_tensorcnt(0);
    }
    __syncthreads();
    if (m0 >= M) return;

    const int col   = lane & 15;
    const int half  = lane >> 4;       // 0 or 1
    const int arow  = m0 + col;        // A row held by this lane
    const int kbA   = half * 8;        // A k-base within a 32-chunk
    const int kbB   = half * 8;        // B k-base within a 32-chunk

    for (int n0 = 0; n0 < Ncol; n0 += 16) {
        const float bv = bias[n0 + col];
        v8f acc = {bv, bv, bv, bv, bv, bv, bv, bv};
        for (int k0 = 0; k0 < K; k0 += 32) {
            const _Float16* ap = A + (size_t)arow * K + k0 + kbA;
            __builtin_prefetch(ap + 32, 0, 0);
            v8h alo = *reinterpret_cast<const v8h*>(ap);
            v8h ahi = *reinterpret_cast<const v8h*>(ap + 16);
            v16h a;
            #pragma unroll
            for (int i = 0; i < 8; ++i) { a[i] = alo[i]; a[8 + i] = ahi[i]; }

            const _Float16* bp = &sB[(n0 + col) * K + k0 + kbB];
            v8h blo = *reinterpret_cast<const v8h*>(bp);
            v8h bhi = *reinterpret_cast<const v8h*>(bp + 16);
            v16h b;
            #pragma unroll
            for (int i = 0; i < 8; ++i) { b[i] = blo[i]; b[8 + i] = bhi[i]; }

            acc = __builtin_amdgcn_wmma_f32_16x16x32_f16(
                false, a, false, b, (short)0, acc, false, false);
        }
        #pragma unroll
        for (int r = 0; r < 8; ++r) {
            int row = m0 + r + half * 8;
            float v = acc[r];
            if (relu) v = v > 0.f ? v : 0.f;
            C[(size_t)row * Ncol + n0 + col] = v;
        }
    }
}

// ---------------- fused edge kernel ----------------
// Per wave: 16 edges.  A[e,k] = relu(a_e*e1W[k]+e1b[k]) built in-register,
// ee = A @ e2 (WMMA, bias in acc; e2 weights staged in LDS via TDM),
// msg = relu(h[src]+ee), atomic-add into agg[dst].
__global__ __launch_bounds__(128) void k_edge_fused(
    const float* __restrict__ eattr,
    const int* __restrict__ src, const int* __restrict__ dst,
    const float* __restrict__ e1W, const float* __restrict__ e1b,
    const _Float16* __restrict__ e2t, const float* __restrict__ e2b,
    const float* __restrict__ h, float* __restrict__ agg)
{
    __shared__ __attribute__((aligned(16))) _Float16 sE[HID * HID];   // 32 KB

    const int lane = threadIdx.x & 31;
    const int wave = threadIdx.x >> 5;
    const int e0 = (blockIdx.x * 4 + wave) * 16;   // E % 64 == 0: exact

    if (wave == 0) {
        tdm_load_2d((unsigned int)(size_t)&sE[0], e2t, HID / 4, HID);
        __builtin_amdgcn_s_wait_tensorcnt(0);
    }
    __syncthreads();

    const int col  = lane & 15;
    const int half = lane >> 4;
    const int kbA  = half * 8;
    const int kbB  = half * 8;

    const float ae = eattr[e0 + col];   // edge scalar for this lane's A-row

    // Build the 4 A fragments (K = 128) once.
    v16h afrag[4];
    #pragma unroll
    for (int kt = 0; kt < 4; ++kt) {
        const int k0 = kt * 32;
        v16h a;
        #pragma unroll
        for (int i = 0; i < 8; ++i) {
            int ka = k0 + kbA + i;
            int kc = ka + 16;
            float v1 = fmaxf(ae * e1W[ka] + e1b[ka], 0.f);
            float v2 = fmaxf(ae * e1W[kc] + e1b[kc], 0.f);
            a[i]     = (_Float16)v1;
            a[8 + i] = (_Float16)v2;
        }
        afrag[kt] = a;
    }

    int sidx[8], didx[8];
    #pragma unroll
    for (int r = 0; r < 8; ++r) {
        int e = e0 + r + half * 8;
        sidx[r] = src[e];
        didx[r] = dst[e];
    }

    for (int n0 = 0; n0 < HID; n0 += 16) {
        const float bv = e2b[n0 + col];
        v8f acc = {bv, bv, bv, bv, bv, bv, bv, bv};
        #pragma unroll
        for (int kt = 0; kt < 4; ++kt) {
            const _Float16* bp = &sE[(n0 + col) * HID + kt * 32 + kbB];
            v8h blo = *reinterpret_cast<const v8h*>(bp);
            v8h bhi = *reinterpret_cast<const v8h*>(bp + 16);
            v16h b;
            #pragma unroll
            for (int i = 0; i < 8; ++i) { b[i] = blo[i]; b[8 + i] = bhi[i]; }
            acc = __builtin_amdgcn_wmma_f32_16x16x32_f16(
                false, afrag[kt], false, b, (short)0, acc, false, false);
        }
        #pragma unroll
        for (int r = 0; r < 8; ++r) {
            float m = acc[r] + h[(size_t)sidx[r] * HID + n0 + col];
            m = m > 0.f ? m : 0.f;
            atomicAdd(&agg[(size_t)didx[r] * HID + n0 + col], m);
        }
    }
}

// ---------------- initial edge-encoder scatter: h_new[idx] = relu(h_old[idx] + a*W + b) ----------------
__global__ __launch_bounds__(256) void k_scatter_set(
    const float* __restrict__ h_old, float* __restrict__ h_new,
    const int* __restrict__ idxArr, const float* __restrict__ eattr,
    const float* __restrict__ encW, const float* __restrict__ encb)
{
    long t = (long)blockIdx.x * blockDim.x + threadIdx.x;
    long e = t >> 5;
    if (e >= EE) return;
    int c = ((int)t & 31) * 4;
    float ae = eattr[e];
    int idx = idxArr[e];
    float4 w  = *reinterpret_cast<const float4*>(encW + c);
    float4 bb = *reinterpret_cast<const float4*>(encb + c);
    float4 hv = *reinterpret_cast<const float4*>(h_old + (size_t)idx * HID + c);
    float4 o;
    o.x = fmaxf(hv.x + ae * w.x + bb.x, 0.f);
    o.y = fmaxf(hv.y + ae * w.y + bb.y, 0.f);
    o.z = fmaxf(hv.z + ae * w.z + bb.z, 0.f);
    o.w = fmaxf(hv.w + ae * w.w + bb.w, 0.f);
    *reinterpret_cast<float4*>(h_new + (size_t)idx * HID + c) = o;
}

// ---------------- combine: zin = f16((1+eps)*h + agg) ----------------
__global__ __launch_bounds__(256) void k_combine(
    const float* __restrict__ h, const float* __restrict__ agg,
    const float* __restrict__ eps_arr, int l, _Float16* __restrict__ zin, long n)
{
    float ep = 1.0f + eps_arr[l];
    long i = (long)blockIdx.x * blockDim.x + threadIdx.x;
    long stride = (long)gridDim.x * blockDim.x;
    for (; i < n; i += stride) zin[i] = (_Float16)(ep * h[i] + agg[i]);
}

// ---------------- BN: column sums / sums of squares ----------------
__global__ __launch_bounds__(256) void k_bn_stats(const float* __restrict__ z,
                                                  float* __restrict__ stats, int rows)
{
    int c = threadIdx.x;                       // 256 columns
    int r0 = blockIdx.x * 256;
    int r1 = r0 + 256; if (r1 > rows) r1 = rows;
    float s = 0.f, s2 = 0.f;
    for (int r = r0; r < r1; ++r) {
        float v = z[(size_t)r * HID2 + c];
        s += v; s2 += v * v;
    }
    atomicAdd(&stats[c], s);
    atomicAdd(&stats[HID2 + c], s2);
}

// ---------------- BN apply + relu + f16 convert ----------------
__global__ __launch_bounds__(256) void k_bn_apply(
    const float* __restrict__ z, const float* __restrict__ stats,
    const float* __restrict__ g, const float* __restrict__ b,
    _Float16* __restrict__ zo, long n)
{
    const float invN = 1.0f / (float)NN;
    long i = (long)blockIdx.x * blockDim.x + threadIdx.x;
    long stride = (long)gridDim.x * blockDim.x;
    for (; i < n; i += stride) {
        int c = (int)(i & (HID2 - 1));
        float mu  = stats[c] * invN;
        float var = stats[HID2 + c] * invN - mu * mu;
        float v = (z[i] - mu) * rsqrtf(var + BN_EPS) * g[c] + b[c];
        zo[i] = (_Float16)(v > 0.f ? v : 0.f);
    }
}

// ---------------- host orchestration ----------------
extern "C" void kernel_launch(void* const* d_in, const int* in_sizes, int n_in,
                              void* d_out, int out_size, void* d_ws, size_t ws_size,
                              hipStream_t stream)
{
    const float* x     = (const float*)d_in[0];
    const int*   eidx  = (const int*)  d_in[1];
    const float* eattr = (const float*)d_in[2];
    const float* lin_W = (const float*)d_in[3];
    const float* lin_b = (const float*)d_in[4];
    const float* enc_W = (const float*)d_in[5];
    const float* enc_b = (const float*)d_in[6];
    const float* eps   = (const float*)d_in[7];
    const float* e1_W  = (const float*)d_in[8];
    const float* e1_b  = (const float*)d_in[9];
    const float* e2_W  = (const float*)d_in[10];
    const float* e2_b  = (const float*)d_in[11];
    const float* m1_W  = (const float*)d_in[12];
    const float* m1_b  = (const float*)d_in[13];
    const float* bn_g  = (const float*)d_in[14];
    const float* bn_b  = (const float*)d_in[15];
    const float* m2_W  = (const float*)d_in[16];
    const float* m2_b  = (const float*)d_in[17];
    const int* src = eidx;
    const int* dst = eidx + EE;

    // workspace carve-out (256B aligned chunks)
    char* wsb = (char*)d_ws;
    size_t off = 0;
    auto carve = [&](size_t bytes) -> char* {
        char* p = wsb + off;
        off += (bytes + 255) & ~(size_t)255;
        return p;
    };
    _Float16* a16   = (_Float16*)carve((size_t)NN * HID  * 2);  // x_f16, reused as zin_f16
    float*    h0    = (float*)   carve((size_t)NN * HID  * 4);
    float*    h1    = (float*)   carve((size_t)NN * HID  * 4);
    float*    agg   = (float*)   carve((size_t)NN * HID  * 4);
    float*    z     = (float*)   carve((size_t)NN * HID2 * 4);
    _Float16* z16   = (_Float16*)carve((size_t)NN * HID2 * 2);
    float*    stats = (float*)   carve(2 * HID2 * 4);
    _Float16* linT  = (_Float16*)carve((size_t)HID * HID * 2);
    _Float16* e2T   = (_Float16*)carve((size_t)NLAY * HID * HID * 2);
    _Float16* m1T   = (_Float16*)carve((size_t)NLAY * HID * HID2 * 2);
    _Float16* m2T   = (_Float16*)carve((size_t)NLAY * HID2 * HID * 2);

    const int TPB = 256;
    const long nh  = (long)NN * HID;
    const long nh2 = (long)NN * HID2;

    // 0. weight conversions (f16, transposed) + x -> f16
    k_cvt_f16<<<4096, TPB, 0, stream>>>(x, a16, nh);
    k_transpose_cvt<<<(HID * HID + TPB - 1) / TPB, TPB, 0, stream>>>(lin_W, linT, HID, HID);
    for (int l = 0; l < NLAY; ++l) {
        k_transpose_cvt<<<(HID * HID + TPB - 1) / TPB, TPB, 0, stream>>>(
            e2_W + (size_t)l * HID * HID, e2T + (size_t)l * HID * HID, HID, HID);
        k_transpose_cvt<<<(HID * HID2 + TPB - 1) / TPB, TPB, 0, stream>>>(
            m1_W + (size_t)l * HID * HID2, m1T + (size_t)l * HID * HID2, HID, HID2);
        k_transpose_cvt<<<(HID2 * HID + TPB - 1) / TPB, TPB, 0, stream>>>(
            m2_W + (size_t)l * HID2 * HID, m2T + (size_t)l * HID2 * HID, HID2, HID);
    }

    // 1. h = x @ lin_W + lin_b
    const int gemm_blocks_N = (NN / 16 + 7) / 8;   // 8 waves/block, 16 rows/wave
    k_gemm_f16<<<gemm_blocks_N, 256, 0, stream>>>(a16, linT, lin_b, h0, NN, HID, HID, 0);

    // 2. indexed-assignment edge encoder (src scatter, then dst scatter overrides)
    hipMemcpyAsync(h1, h0, nh * 4, hipMemcpyDeviceToDevice, stream);
    {
        long t = (long)EE * 32;
        int blocks = (int)((t + TPB - 1) / TPB);
        k_scatter_set<<<blocks, TPB, 0, stream>>>(h0, h1, src, eattr, enc_W, enc_b);
        k_scatter_set<<<blocks, TPB, 0, stream>>>(h0, h1, dst, eattr, enc_W, enc_b);
    }

    float* h_cur  = h1;
    float* h_next = h0;

    // 3. GIN layers
    for (int l = 0; l < NLAY; ++l) {
        k_zero<<<4096, TPB, 0, stream>>>(agg, nh);
        k_edge_fused<<<EE / 64, 128, 0, stream>>>(
            eattr, src, dst,
            e1_W + (size_t)l * HID, e1_b + (size_t)l * HID,
            e2T + (size_t)l * HID * HID, e2_b + (size_t)l * HID,
            h_cur, agg);
        k_combine<<<4096, TPB, 0, stream>>>(h_cur, agg, eps, l, a16, nh);
        k_gemm_f16<<<gemm_blocks_N, 256, 0, stream>>>(
            a16, m1T + (size_t)l * HID * HID2, m1_b + (size_t)l * HID2, z,
            NN, HID, HID2, 0);
        k_zero<<<2, TPB, 0, stream>>>(stats, 2 * HID2);
        k_bn_stats<<<(NN + 255) / 256, 256, 0, stream>>>(z, stats, NN);
        k_bn_apply<<<4096, TPB, 0, stream>>>(
            z, stats, bn_g + (size_t)l * HID2, bn_b + (size_t)l * HID2, z16, nh2);
        float* outp = (l == NLAY - 1) ? (float*)d_out : h_next;
        k_gemm_f16<<<gemm_blocks_N, 256, 0, stream>>>(
            z16, m2T + (size_t)l * HID2 * HID, m2_b + (size_t)l * HID, outp,
            NN, HID2, HID, (l < NLAY - 1) ? 1 : 0);
        float* tmp = h_cur; h_cur = h_next; h_next = tmp;
    }

    (void)in_sizes; (void)n_in; (void)out_size; (void)ws_size;
}
